// ResBlock_A_15814069584193
// MI455X (gfx1250) — matrix operand order repaired and checked by reading, not compiled, not run on hardware
//
#include <hip/hip_runtime.h>

// ---------------------------------------------------------------------------
// Problem constants (match reference setup_inputs)
// ---------------------------------------------------------------------------
#define NPTS 200000
#define MPTS 50000
#define KNBR 32
#define PKP  15
#define IND  128
#define OUTD 256
#define HIDD 64
#define EPSBN 1e-5f
#define SLOPE 0.1f
#define KPEXT 0.05f

typedef __attribute__((ext_vector_type(16))) _Float16 v16h;
typedef __attribute__((ext_vector_type(8)))  _Float16 v8h;
typedef __attribute__((ext_vector_type(8)))  float    v8f;

__device__ __forceinline__ float lrelu(float x) { return x > 0.f ? x : SLOPE * x; }

// ---------------------------------------------------------------------------
// Pack a row-major f32 weight matrix B[K,N] into per-lane WMMA B-fragment
// order (f16): Bp[tileN][chunk][lane][j] = B[chunk*32 + kb + (j<8 ? j : j+8), tileN*16 + n]
// with kb = (lane<16)?0:8, n = lane&15.  Each GEMM lane then reads its 16
// halves as one contiguous 32-byte vector.
// ---------------------------------------------------------------------------
__global__ void pack_b_f16(const float* __restrict__ B, _Float16* __restrict__ Bp,
                           int Kdim, int Ncols) {
    const int  nChunks = Kdim >> 5;
    const long total   = (long)(Ncols >> 4) * nChunks * 32 * 16;
    long i = (long)blockIdx.x * blockDim.x + threadIdx.x;
    const long stride = (long)gridDim.x * blockDim.x;
    for (; i < total; i += stride) {
        const int  j     = (int)(i & 15);
        const int  lane  = (int)((i >> 4) & 31);
        const long rest  = i >> 9;
        const int  chunk = (int)(rest % nChunks);
        const int  tileN = (int)(rest / nChunks);
        const int  kb    = (lane < 16) ? 0 : 8;
        const int  n     = tileN * 16 + (lane & 15);
        const int  k     = chunk * 32 + kb + (j < 8 ? j : j + 8);
        Bp[i] = (_Float16)B[(long)k * Ncols + n];
    }
}

// ---------------------------------------------------------------------------
// WMMA GEMM: C[M,N] = A[M,K](f16,row-major) * B(packed f16) + bias, fp32 acc.
// Block = 4 waves, wave w owns rows [blockIdx.y*64 + w*16, +16), cols
// [blockIdx.x*32, +32) (two 16x16 tiles sharing the A fragment).
// A fragment: two contiguous 8-half (16B) loads per chunk.
// B fragment: one contiguous 32B load per tile per chunk.
// Requires Mrows % 16 == 0 (true for 200000 / 50000), Kdim % 32 == 0.
// ---------------------------------------------------------------------------
__global__ void gemm_wmma_f16(const _Float16* __restrict__ A,
                              const v16h* __restrict__ Bp,
                              const float* __restrict__ bias,
                              float* __restrict__ C,
                              int Mrows, int Kdim, int lda, int ldc) {
    const int lane  = threadIdx.x & 31;
    const int wave  = threadIdx.x >> 5;
    const int tileM = blockIdx.y * 4 + wave;
    const int row0  = tileM * 16;
    if (row0 >= Mrows) return;                 // wave-uniform exit
    const int col0    = blockIdx.x * 32;
    const int nChunks = Kdim >> 5;

    // A: lane<16 -> M=lane, K in {0..7,16..23}; lane>=16 -> M=lane-16, +8
    const int mA  = lane & 15;
    const int kbA = (lane < 16) ? 0 : 8;
    const _Float16* Arow = A + (size_t)(row0 + mA) * lda + kbA;

    const v16h* B0 = Bp + ((size_t)(col0 >> 4) * nChunks) * 32 + lane;
    const v16h* B1 = B0 + (size_t)nChunks * 32;

    v8f acc0 = {}, acc1 = {};
#pragma unroll 2
    for (int c = 0; c < nChunks; ++c) {
        const v8h alo = *(const v8h*)(Arow + c * 32);        // K kb..kb+7
        const v8h ahi = *(const v8h*)(Arow + c * 32 + 16);   // K kb+16..kb+23
        const v16h a = __builtin_shufflevector(alo, ahi,
                          0, 1, 2, 3, 4, 5, 6, 7, 8, 9, 10, 11, 12, 13, 14, 15);
        const v16h b0 = B0[(size_t)c * 32];
        const v16h b1 = B1[(size_t)c * 32];
        acc0 = __builtin_amdgcn_wmma_f32_16x16x32_f16(
                   false, a, false, b0, (short)0, acc0, false, false);
        acc1 = __builtin_amdgcn_wmma_f32_16x16x32_f16(
                   false, a, false, b1, (short)0, acc1, false, false);
    }

    // C/D layout: VGPR r -> row (r or 8+r), N = lane&15
    const int mb = (lane < 16) ? 0 : 8;
    const int nC = lane & 15;
    float* Crow = C + (size_t)(row0 + mb) * ldc + col0 + nC;
    const float bv0 = bias[col0 + nC];
    const float bv1 = bias[col0 + 16 + nC];
#pragma unroll
    for (int r = 0; r < 8; ++r) {
        Crow[(size_t)r * ldc]      = acc0[r] + bv0;
        Crow[(size_t)r * ldc + 16] = acc1[r] + bv1;
    }
}

// ---------------------------------------------------------------------------
// Elementwise helpers
// ---------------------------------------------------------------------------
__global__ void cvt_f32_f16(const float* __restrict__ x, _Float16* __restrict__ y, long n) {
    long i = (long)blockIdx.x * blockDim.x + threadIdx.x;
    const long stride = (long)gridDim.x * blockDim.x;
    for (; i < n; i += stride) y[i] = (_Float16)x[i];
}

__global__ void zero_f32(float* __restrict__ p, int n) {
    int i = blockIdx.x * blockDim.x + threadIdx.x;
    if (i < n) p[i] = 0.f;
}

__global__ void lrelu_f32_to_f16(const float* __restrict__ x, _Float16* __restrict__ y, long n) {
    long i = (long)blockIdx.x * blockDim.x + threadIdx.x;
    const long stride = (long)gridDim.x * blockDim.x;
    for (; i < n; i += stride) y[i] = (_Float16)lrelu(x[i]);
}

// Column sums + sums-of-squares; blockDim.x == cols, one atomic pair per
// column per 256-row block.
__global__ void col_stats(const float* __restrict__ X, float* __restrict__ stats,
                          int rows, int cols) {
    const int c = threadIdx.x;
    long r0 = (long)blockIdx.x * 256;
    long r1 = r0 + 256; if (r1 > rows) r1 = rows;
    float s = 0.f, s2 = 0.f;
    for (long r = r0; r < r1; ++r) {
        const float v = X[r * cols + c];
        s += v; s2 += v * v;
    }
    atomicAdd(&stats[c], s);
    atomicAdd(&stats[cols + c], s2);
}

// BN (population stats) + LeakyReLU, emit f16 for the next WMMA stage.
// cols must be a power of two.
__global__ void bn_lrelu_f16(const float* __restrict__ X, _Float16* __restrict__ Y,
                             const float* __restrict__ stats,
                             const float* __restrict__ g, const float* __restrict__ be,
                             long rows, int cols, float invRows) {
    const long n = rows * (long)cols;
    long i = (long)blockIdx.x * blockDim.x + threadIdx.x;
    const long stride = (long)gridDim.x * blockDim.x;
    for (; i < n; i += stride) {
        const int c = (int)(i & (cols - 1));
        const float mu  = stats[c] * invRows;
        const float var = stats[cols + c] * invRows - mu * mu;
        const float rs  = rsqrtf(var + EPSBN);
        Y[i] = (_Float16)lrelu((X[i] - mu) * rs * g[c] + be[c]);
    }
}

// ---------------------------------------------------------------------------
// KPConv neighbor aggregation: agg[m,p,c] = sum_k w[m,k,p] * x1[pools[m,k],c]
// One query per 64-thread block (thread = channel).  Gathers of x1h (25.6 MB
// f16) hit the 192 MB global L2.
// ---------------------------------------------------------------------------
__global__ void kpconv_agg(const float* __restrict__ points,
                           const float* __restrict__ subp,
                           const int* __restrict__ pools,
                           const _Float16* __restrict__ x1h,
                           const float* __restrict__ kp,
                           _Float16* __restrict__ aggh) {
    __shared__ int   idx_s[KNBR];
    __shared__ float rel[KNBR][3];
    __shared__ float w_s[KNBR][16];   // 15 used, padded

    const int m = blockIdx.x;
    const int t = threadIdx.x;        // 0..63

    if (t < KNBR) {
        const int idx = pools[(long)m * KNBR + t];
        idx_s[t] = idx;
        rel[t][0] = points[(long)idx * 3 + 0] - subp[(long)m * 3 + 0];
        rel[t][1] = points[(long)idx * 3 + 1] - subp[(long)m * 3 + 1];
        rel[t][2] = points[(long)idx * 3 + 2] - subp[(long)m * 3 + 2];
    }
    __syncthreads();

    for (int t2 = t; t2 < KNBR * PKP; t2 += 64) {
        const int k = t2 / PKP, p = t2 % PKP;
        const float dx = rel[k][0] - kp[p * 3 + 0];
        const float dy = rel[k][1] - kp[p * 3 + 1];
        const float dz = rel[k][2] - kp[p * 3 + 2];
        const float dist = sqrtf(dx * dx + dy * dy + dz * dz);
        w_s[k][p] = fmaxf(0.f, 1.f - dist * (1.f / KPEXT));
    }
    __syncthreads();

    const int c = t;
    float acc[PKP];
#pragma unroll
    for (int p = 0; p < PKP; ++p) acc[p] = 0.f;

    for (int k = 0; k < KNBR; ++k) {
        const float f = (float)x1h[(long)idx_s[k] * HIDD + c];
#pragma unroll
        for (int p = 0; p < PKP; ++p) acc[p] += w_s[k][p] * f;
    }

    _Float16* dst = aggh + (long)m * (PKP * HIDD);
#pragma unroll
    for (int p = 0; p < PKP; ++p) dst[p * HIDD + c] = (_Float16)acc[p];
}

// Shortcut: max over K neighbors of (f16-staged) features. thread = channel.
__global__ void scmax_f16(const _Float16* __restrict__ xh,
                          const int* __restrict__ pools,
                          _Float16* __restrict__ out) {
    __shared__ int idx_s[KNBR];
    const int m = blockIdx.x;
    const int t = threadIdx.x;        // 0..127
    if (t < KNBR) idx_s[t] = pools[(long)m * KNBR + t];
    __syncthreads();
    float mx = -3.0e38f;
    for (int k = 0; k < KNBR; ++k)
        mx = fmaxf(mx, (float)xh[(long)idx_s[k] * IND + t]);
    out[(long)m * IND + t] = (_Float16)mx;
}

// Final: lrelu( lrelu(BN2(x2)) + lrelu(BNs(sc2)) ) -> fp32 output.
__global__ void final_combine(const float* __restrict__ X2, const float* __restrict__ SC,
                              const float* __restrict__ st2, const float* __restrict__ sts,
                              const float* __restrict__ g2, const float* __restrict__ be2,
                              const float* __restrict__ gs, const float* __restrict__ bes,
                              float* __restrict__ out, long rows, int cols, float invRows) {
    const long n = rows * (long)cols;
    long i = (long)blockIdx.x * blockDim.x + threadIdx.x;
    const long stride = (long)gridDim.x * blockDim.x;
    for (; i < n; i += stride) {
        const int c = (int)(i & (cols - 1));
        float mu  = st2[c] * invRows;
        float var = st2[cols + c] * invRows - mu * mu;
        const float a = lrelu((X2[i] - mu) * rsqrtf(var + EPSBN) * g2[c] + be2[c]);
        mu  = sts[c] * invRows;
        var = sts[cols + c] * invRows - mu * mu;
        const float b = lrelu((SC[i] - mu) * rsqrtf(var + EPSBN) * gs[c] + bes[c]);
        out[i] = lrelu(a + b);
    }
}

// ---------------------------------------------------------------------------
// Host orchestration
// ---------------------------------------------------------------------------
extern "C" void kernel_launch(void* const* d_in, const int* in_sizes, int n_in,
                              void* d_out, int out_size, void* d_ws, size_t ws_size,
                              hipStream_t stream) {
    (void)in_sizes; (void)n_in; (void)out_size; (void)ws_size;

    const float* points = (const float*)d_in[0];
    const float* subp   = (const float*)d_in[1];
    const float* feats  = (const float*)d_in[2];
    const int*   pools  = (const int*)d_in[3];
    const float* kp     = (const float*)d_in[4];
    const float* W1  = (const float*)d_in[5];
    const float* b1  = (const float*)d_in[6];
    const float* g1  = (const float*)d_in[7];
    const float* be1 = (const float*)d_in[8];
    const float* kpw = (const float*)d_in[9];
    const float* W2  = (const float*)d_in[10];
    const float* b2  = (const float*)d_in[11];
    const float* g2  = (const float*)d_in[12];
    const float* be2 = (const float*)d_in[13];
    const float* Ws  = (const float*)d_in[14];
    const float* bs  = (const float*)d_in[15];
    const float* gs  = (const float*)d_in[16];
    const float* bes = (const float*)d_in[17];
    float* out = (float*)d_out;

    // Workspace carve-out (256B aligned slices)
    char* wp = (char*)d_ws;
    auto carve = [&](size_t bytes) -> char* {
        char* p = wp;
        wp += (bytes + 255) & ~(size_t)255;
        return p;
    };
    _Float16* Xh   = (_Float16*)carve((size_t)NPTS * IND * 2);         // f16 features
    _Float16* Bp1  = (_Float16*)carve((size_t)IND * HIDD * 2);         // packed W1
    _Float16* Bp2  = (_Float16*)carve((size_t)HIDD * OUTD * 2);        // packed W2
    _Float16* Bps  = (_Float16*)carve((size_t)IND * OUTD * 2);         // packed Ws
    _Float16* Bpk  = (_Float16*)carve((size_t)PKP * HIDD * HIDD * 2);  // packed kp_w [960,64]
    float*    X1   = (float*)   carve((size_t)NPTS * HIDD * 4);        // conv1 raw
    _Float16* X1h  = (_Float16*)carve((size_t)NPTS * HIDD * 2);        // post BN+lrelu
    _Float16* AGGh = (_Float16*)carve((size_t)MPTS * PKP * HIDD * 2);  // [M,960]
    float*    XKP  = (float*)   carve((size_t)MPTS * HIDD * 4);
    _Float16* XKPh = (_Float16*)carve((size_t)MPTS * HIDD * 2);
    float*    X2   = (float*)   carve((size_t)MPTS * OUTD * 4);
    _Float16* SCh  = (_Float16*)carve((size_t)MPTS * IND * 2);
    float*    SC2  = (float*)   carve((size_t)MPTS * OUTD * 4);
    // stats + zero-bias: contiguous region zeroed every call
    float*    st1  = (float*)   carve((size_t)2 * HIDD * 4);           // 512 B
    float*    st2  = (float*)   carve((size_t)2 * OUTD * 4);           // 2048 B
    float*    st3  = (float*)   carve((size_t)2 * OUTD * 4);           // 2048 B
    float*    zb   = (float*)   carve((size_t)OUTD * 4);               // 1024 B
    (void)zb;

    const int thr = 256;
    const int nZero = 2 * HIDD + 4 * OUTD + OUTD;                      // 1408 floats

    // Stage operands: features -> f16; weights -> packed f16 B-fragments
    cvt_f32_f16<<<dim3(2048), thr, 0, stream>>>(feats, Xh, (long)NPTS * IND);
    pack_b_f16<<<dim3(16),  thr, 0, stream>>>(W1,  Bp1, IND,        HIDD);
    pack_b_f16<<<dim3(32),  thr, 0, stream>>>(W2,  Bp2, HIDD,       OUTD);
    pack_b_f16<<<dim3(64),  thr, 0, stream>>>(Ws,  Bps, IND,        OUTD);
    pack_b_f16<<<dim3(120), thr, 0, stream>>>(kpw, Bpk, PKP * HIDD, HIDD);
    zero_f32<<<dim3(3), 512, 0, stream>>>(st1, nZero);

    // Conv1: [N,128]x[128,64]+b1 -> X1 ; BN stats over N ; BN+lrelu -> f16
    gemm_wmma_f16<<<dim3(HIDD / 32, (NPTS + 63) / 64), 128, 0, stream>>>(
        Xh, (const v16h*)Bp1, b1, X1, NPTS, IND, IND, HIDD);
    col_stats<<<dim3((NPTS + 255) / 256), HIDD, 0, stream>>>(X1, st1, NPTS, HIDD);
    bn_lrelu_f16<<<dim3(2048), thr, 0, stream>>>(X1, X1h, st1, g1, be1,
                                                 NPTS, HIDD, 1.f / NPTS);

    // KPConv: irregular aggregation -> [M,960] f16, then WMMA GEMM with kp_w
    kpconv_agg<<<dim3(MPTS), 64, 0, stream>>>(points, subp, pools, X1h, kp, AGGh);
    gemm_wmma_f16<<<dim3(HIDD / 32, (MPTS + 63) / 64), 128, 0, stream>>>(
        AGGh, (const v16h*)Bpk, zb, XKP, MPTS, PKP * HIDD, PKP * HIDD, HIDD);
    lrelu_f32_to_f16<<<dim3(1024), thr, 0, stream>>>(XKP, XKPh, (long)MPTS * HIDD);

    // Conv2: [M,64]x[64,256]+b2 -> X2 ; stats over M
    gemm_wmma_f16<<<dim3(OUTD / 32, (MPTS + 63) / 64), 128, 0, stream>>>(
        XKPh, (const v16h*)Bp2, b2, X2, MPTS, HIDD, HIDD, OUTD);
    col_stats<<<dim3((MPTS + 255) / 256), OUTD, 0, stream>>>(X2, st2, MPTS, OUTD);

    // Shortcut: neighbor max -> [M,128] f16 ; GEMM with Ws ; stats
    scmax_f16<<<dim3(MPTS), IND, 0, stream>>>(Xh, pools, SCh);
    gemm_wmma_f16<<<dim3(OUTD / 32, (MPTS + 63) / 64), 128, 0, stream>>>(
        SCh, (const v16h*)Bps, bs, SC2, MPTS, IND, IND, OUTD);
    col_stats<<<dim3((MPTS + 255) / 256), OUTD, 0, stream>>>(SC2, st3, MPTS, OUTD);

    // Fused BN+lrelu on both branches, add, final lrelu -> fp32 out
    final_combine<<<dim3(4096), thr, 0, stream>>>(
        X2, SC2, st2, st3, g2, be2, gs, bes, out, MPTS, OUTD, 1.f / MPTS);
}
